// Half_Graph_2714419331162
// MI455X (gfx1250) — matrix operand order repaired and checked by reading, not compiled, not run on hardware
//
#include <hip/hip_runtime.h>

// ---------------------------------------------------------------------------
// MI455X (gfx1250) fused Half_Graph kernel.
// Memory-bound problem (~142 MB min traffic, AI ~25 FLOP/B << bf16-WMMA
// crossover) -> single fused pass; all 1x1 convs as v_wmma_f32_16x16x32_bf16
// with BN folded into weights and biases injected via the C accumulator.
// Round 3: single-instruction ReLU/clamp via v_med3_f32 (fmed3 builtin).
// ---------------------------------------------------------------------------

#define HD    10
#define C2    20
#define HWSZ  (192*192)          // 36864
#define NB    8
#define NPIX  (NB*HWSZ)          // 294912
#define NTILES (NPIX/16)         // 18432  (16 pixels per wave-tile)

#define FRAG_SLOTS 15
#define FRAG_DW   (FRAG_SLOTS*256)      // bf16 A-fragments, lane-major dwords
#define BIAS_VECS 10
#define BIAS_FLOATS (BIAS_VECS*32)
#define WS_DW     (FRAG_DW + BIAS_FLOATS)   // 4160 dwords in workspace

#define SCR_ROWS   48
#define DUMP_ROW   47                    // branchless sink for invalid stores
#define SCR_FLOATS (SCR_ROWS*16)         // per-wave scratch floats
#define WAVES_PER_BLOCK 8
#define LDS_DW (WS_DW + WAVES_PER_BLOCK*SCR_FLOATS)

typedef __attribute__((ext_vector_type(16))) __bf16 v16bf;
typedef __attribute__((ext_vector_type(8)))  float  v8f;

union Frag {
  v16bf v;
  __bf16 b[16];
  unsigned u[8];
  uint4 q[2];
};

__device__ __forceinline__ unsigned short f2bf(float f){
  union { float f; unsigned u; } cv; cv.f = f;
  unsigned r = cv.u + 0x7FFFu + ((cv.u >> 16) & 1u);   // RNE truncate
  return (unsigned short)(r >> 16);
}
__device__ __forceinline__ unsigned pack2bf(float a, float b){
  return (unsigned)f2bf(a) | ((unsigned)f2bf(b) << 16);
}

// ---------------------------------------------------------------------------
// Prep: fold BN into weights, emit bf16 A-fragments (16x32 A layout,
// lane-major: dword index = slot*256 + lane*8 + j) and padded bias vectors.
// A order used here: lanes0-15: V0..7 = K0..15 pairs;
//           lanes16-31: V0..7 = K16..31 pairs (k = 16*hi + 2j).
// ---------------------------------------------------------------------------
__global__ void prep_weights(
    const float* dW1, const float* dbn1, const float* dW2, const float* dbn2,
    const float* uW1, const float* ubn1, const float* uW2, const float* ubn2,
    const float* lW1, const float* lbn1, const float* lW2, const float* lbn2,
    const float* guWg, const float* gubg, const float* guWc, const float* gubc,
    const float* glWg, const float* glbg, const float* glWc, const float* glbc,
    unsigned* ws)
{
  int idx = blockIdx.x * blockDim.x + threadIdx.x;
  if (idx < FRAG_DW) {
    int slot = idx >> 8;
    int r = idx & 255;
    int L = r >> 3, j = r & 7;
    int hi = L >> 4, m = L & 15;
    int k0 = hi*16 + 2*j;
    const float* W; const float* bn; int Cout; int mb;
    switch (slot) {
      case 0:  W=dW1;  bn=dbn1;    Cout=20; mb=0;  break;
      case 1:  W=dW1;  bn=dbn1;    Cout=20; mb=16; break;
      case 2:  W=dW2;  bn=dbn2;    Cout=10; mb=0;  break;
      case 3:  W=uW1;  bn=ubn1;    Cout=20; mb=0;  break;
      case 4:  W=uW1;  bn=ubn1;    Cout=20; mb=16; break;
      case 5:  W=uW2;  bn=ubn2;    Cout=10; mb=0;  break;
      case 6:  W=lW1;  bn=lbn1;    Cout=20; mb=0;  break;
      case 7:  W=lW1;  bn=lbn1;    Cout=20; mb=16; break;
      case 8:  W=lW2;  bn=lbn2;    Cout=10; mb=0;  break;
      case 9:  W=guWg; bn=nullptr; Cout=20; mb=0;  break;
      case 10: W=guWg; bn=nullptr; Cout=20; mb=16; break;
      case 11: W=guWc; bn=nullptr; Cout=10; mb=0;  break;
      case 12: W=glWg; bn=nullptr; Cout=20; mb=0;  break;
      case 13: W=glWg; bn=nullptr; Cout=20; mb=16; break;
      default: W=glWc; bn=nullptr; Cout=10; mb=0;  break;
    }
    int mg = mb + m;
    float w0 = 0.f, w1 = 0.f;
    if (mg < Cout) {
      float s = 1.f;
      if (bn) s = bn[mg] * rsqrtf(bn[3*Cout + mg] + 1e-5f);
      if (k0     < C2) w0 = s * W[mg*C2 + k0];
      if (k0 + 1 < C2) w1 = s * W[mg*C2 + k0 + 1];
    }
    ws[idx] = pack2bf(w0, w1);
  } else if (idx < WS_DW) {
    int b = idx - FRAG_DW;
    int vec = b >> 5, o = b & 31;
    float* wsb = (float*)ws;
    const float* bn = nullptr; const float* gb = nullptr; int C;
    switch (vec) {
      case 0: bn=dbn1; C=20; break;
      case 1: bn=dbn2; C=10; break;
      case 2: bn=ubn1; C=20; break;
      case 3: bn=ubn2; C=10; break;
      case 4: bn=lbn1; C=20; break;
      case 5: bn=lbn2; C=10; break;
      case 6: gb=gubg; C=20; break;
      case 7: gb=gubc; C=10; break;
      case 8: gb=glbg; C=20; break;
      default: gb=glbc; C=10; break;
    }
    float v = 0.f;
    if (o < C) {
      if (bn) {
        float s = bn[o] * rsqrtf(bn[3*C + o] + 1e-5f);
        v = bn[C + o] - bn[2*C + o] * s;
      } else v = gb[o];
    }
    wsb[idx] = v;
  }
}

// ---------------------------------------------------------------------------
// Main kernel helpers
// ---------------------------------------------------------------------------
__device__ __forceinline__ Frag loadA(const unsigned* fragL, int slot, int lane){
  Frag f;
  const uint4* p = (const uint4*)(fragL + slot*256 + lane*8);
  f.q[0] = p[0];
  f.q[1] = p[1];
  return f;
}

// 8 consecutive bias floats per lane -> two ds_load_b128
__device__ __forceinline__ v8f biasC(const float* biasL, int vec, int base, int hi){
  union { v8f v; float4 f4[2]; } c;
  const float4* p = (const float4*)(biasL + vec*32 + base + hi*8);
  c.f4[0] = p[0];
  c.f4[1] = p[1];
  return c.v;
}

__device__ __forceinline__ v8f wmma_bf(const Frag& a, const Frag& b, v8f c){
  return __builtin_amdgcn_wmma_f32_16x16x32_bf16(false, a.v, false, b.v,
                                                 (short)0, c, false, false);
}

// single v_med3_f32 ReLU: med3(x, 0, +inf)
__device__ __forceinline__ v8f vrelu(v8f x){
#pragma unroll
  for (int r = 0; r < 8; ++r)
    x[r] = __builtin_amdgcn_fmed3f(x[r], 0.f, __builtin_inff());
  return x;
}

__device__ __forceinline__ float sigmoidf(float x){
  return __builtin_amdgcn_rcpf(1.f + __expf(-x));
}
__device__ __forceinline__ float tanh_fast(float x){
  x = __builtin_amdgcn_fmed3f(x, -15.f, 15.f);   // clamp in one v_med3_f32
  float e = __expf(2.f * x);
  return (e - 1.f) * __builtin_amdgcn_rcpf(e + 1.f);
}

// B order used here (contiguous k, not the ISA table order): lanes0-15: N=lane, K pairs 0..15 in V0..7;
//                          lanes16-31: N=lane-16, K pairs 16..31.
// Branchless build of cat(s1*m1, s2*m2): pair (k,k+1) never straddles the
// channel-10 / channel-20 boundaries (k even), so a single pointer/channel/
// multiplier select per pair suffices (lowers to v_cndmask on a lane mask).
__device__ __forceinline__ Frag buildB_gg(const float* s1, float m1,
                                          const float* s2, float m2, int q, int hi){
  Frag f;
#pragma unroll
  for (int j = 0; j < 8; ++j){
    int k = hi*16 + 2*j;
    bool useS1 = (k < HD);
    bool valid = (k < C2);
    const float* sp = useS1 ? s1 : s2;
    int ch = useS1 ? k : (k - HD);
    ch = valid ? ch : 0;
    float mm = useS1 ? m1 : m2;
    float v0 = sp[ch*HWSZ + q];
    float v1 = sp[(ch + 1)*HWSZ + q];
    v0 = valid ? v0 * mm : 0.f;
    v1 = valid ? v1 * mm : 0.f;
    f.b[2*j]   = (__bf16)v0;
    f.b[2*j+1] = (__bf16)v1;
  }
  return f;
}

// Branchless build from scratch rows 0..19 (zeros for k>=20)
__device__ __forceinline__ Frag buildB_scr(const float* scr, int q, int hi){
  Frag f;
#pragma unroll
  for (int j = 0; j < 8; ++j){
    int k = hi*16 + 2*j;
    int k0 = (k     < C2) ? k     : 0;
    int k1 = (k + 1 < C2) ? k + 1 : 0;
    float v0 = scr[k0*16 + q];
    float v1 = scr[k1*16 + q];
    v0 = (k     < C2) ? v0 : 0.f;
    v1 = (k + 1 < C2) ? v1 : 0.f;
    f.b[2*j]   = (__bf16)v0;
    f.b[2*j+1] = (__bf16)v1;
  }
  return f;
}

// D layout: V r <-> m = r + hi*8. Store a 20-channel (two-tile) result as
// [channel][pixel] rows in per-wave LDS scratch; invalid channels go to the
// dump row (unconditional ds_store, no exec juggling).
__device__ __forceinline__ void store20(float* scr, int rowbase, v8f d0, v8f d1, int q, int hi){
#pragma unroll
  for (int r = 0; r < 8; ++r){
    int ch = hi*8 + r;
    scr[(rowbase + ch)*16 + q] = d0[r];
    int ch2 = 16 + hi*8 + r;
    int row2 = (ch2 < C2) ? (rowbase + ch2) : DUMP_ROW;
    scr[row2*16 + q] = d1[r];
  }
}

// Conv1x1(20->20)+BN+ReLU -> Conv1x1(20->10)+BN+ReLU, biases via C accumulator
__device__ __forceinline__ v8f applyBlock(const unsigned* fragL, const float* biasL,
                                          float* scr, const Frag& Bin,
                                          int sA0, int sA1, int sA2,
                                          int vb1, int vb2, int lane){
  int q = lane & 15, hi = lane >> 4;
  Frag A0 = loadA(fragL, sA0, lane);
  Frag A1 = loadA(fragL, sA1, lane);
  v8f h0 = vrelu(wmma_bf(A0, Bin, biasC(biasL, vb1, 0,  hi)));
  v8f h1 = vrelu(wmma_bf(A1, Bin, biasC(biasL, vb1, 16, hi)));
  store20(scr, 0, h0, h1, q, hi);
  Frag B2 = buildB_scr(scr, q, hi);
  Frag A2 = loadA(fragL, sA2, lane);
  return vrelu(wmma_bf(A2, B2, biasC(biasL, vb2, 0, hi)));
}

// ConvGRU cell. Scratch layout: msg rows 0..9, h rows 10..19,
// gates rows 20..39 (r: 20..29, u: 30..39), dump row 47.
__device__ __forceinline__ void gruCell(const unsigned* fragL, const float* biasL,
                                        float* scr, v8f msg, const float* hp,
                                        int sG0, int sG1, int sC, int vbg, int vbc,
                                        float* outp, int lane){
  int q = lane & 15, hi = lane >> 4;
#pragma unroll
  for (int r = 0; r < 8; ++r){
    int ch = hi*8 + r;
    int rowm = (ch < HD) ? ch : DUMP_ROW;
    scr[rowm*16 + q] = msg[r];
    int chh  = (ch < HD) ? ch : 0;
    int rowh = (ch < HD) ? (10 + ch) : DUMP_ROW;
    scr[rowh*16 + q] = hp[chh*HWSZ + q];
  }
  // gate input = cat(msg, h) = scratch rows 0..19
  Frag Bg = buildB_scr(scr, q, hi);
  Frag G0 = loadA(fragL, sG0, lane);
  Frag G1 = loadA(fragL, sG1, lane);
  v8f g0 = wmma_bf(G0, Bg, biasC(biasL, vbg, 0,  hi));
  v8f g1 = wmma_bf(G1, Bg, biasC(biasL, vbg, 16, hi));
#pragma unroll
  for (int r = 0; r < 8; ++r){ g0[r] = sigmoidf(g0[r]); g1[r] = sigmoidf(g1[r]); }
  store20(scr, 20, g0, g1, q, hi);
  // candidate input = cat(msg, r*h): k<10 -> row k ; 10..19 -> row k (h) * row k+10 (r)
  Frag Bc;
#pragma unroll
  for (int j = 0; j < 8; ++j){
    int k = hi*16 + 2*j;
    int a0r = (k     < C2) ? k       : 0;
    int a1r = (k + 1 < C2) ? (k + 1) : 0;
    int b0r = (k     < C2) ? (k + 10) : 0;
    int b1r = (k + 1 < C2) ? (k + 11) : 0;
    float a0 = scr[a0r*16 + q];
    float a1 = scr[a1r*16 + q];
    float b0 = scr[b0r*16 + q];
    float b1 = scr[b1r*16 + q];
    float m0 = (k     < HD) ? 1.f : b0;
    float m1 = (k + 1 < HD) ? 1.f : b1;
    float v0 = (k     < C2) ? a0 * m0 : 0.f;
    float v1 = (k + 1 < C2) ? a1 * m1 : 0.f;
    Bc.b[2*j]   = (__bf16)v0;
    Bc.b[2*j+1] = (__bf16)v1;
  }
  Frag Ac = loadA(fragL, sC, lane);
  v8f c = wmma_bf(Ac, Bc, biasC(biasL, vbc, 0, hi));
#pragma unroll
  for (int r = 0; r < 8; ++r) c[r] = tanh_fast(c[r]);
  // xh' = (1-u)*h + u*c ; u rows 30..39, h rows 10..19
  if (hi == 0){
#pragma unroll
    for (int r = 0; r < 8; ++r){
      float u = scr[(30 + r)*16 + q];
      float h = scr[(10 + r)*16 + q];
      outp[r*HWSZ + q] = (1.f - u)*h + u*c[r];
    }
  } else {
#pragma unroll
    for (int r = 0; r < 2; ++r){
      int ch = 8 + r;
      float u = scr[(30 + ch)*16 + q];
      float h = scr[(10 + ch)*16 + q];
      outp[ch*HWSZ + q] = (1.f - u)*h + u*c[r];
    }
  }
}

// ---------------------------------------------------------------------------
// Main fused kernel: 8 waves/block, each wave grid-strides over 16-pixel tiles
// ---------------------------------------------------------------------------
__global__ __launch_bounds__(256, 1)
void half_graph_main(const float* __restrict__ xf, const float* __restrict__ xh,
                     const float* __restrict__ xp, const float* __restrict__ h_att,
                     const float* __restrict__ p_att, const unsigned* __restrict__ ws,
                     float* __restrict__ out)
{
  __shared__ __align__(16) unsigned lds[LDS_DW];
  for (int i = threadIdx.x; i < WS_DW; i += 256) lds[i] = ws[i];
  __syncthreads();

  const unsigned* fragL = lds;
  const float* biasL = (const float*)(lds + FRAG_DW);
  int lane = threadIdx.x & 31;
  int wave = threadIdx.x >> 5;
  float* scr = (float*)(lds + WS_DW) + wave * SCR_FLOATS;
  int q = lane & 15, hi = lane >> 4;

  int wgid = blockIdx.x * WAVES_PER_BLOCK + wave;
  int nw = gridDim.x * WAVES_PER_BLOCK;
  for (int t = wgid; t < NTILES; t += nw){
    int P  = t * 16;
    int b  = P / HWSZ;
    int hw = P - b*HWSZ;              // tile never crosses batch (HWSZ%16==0)
    const float* xfp  = xf + (b)*HD*HWSZ + hw;
    const float* xh0p = xh + (0*NB + b)*HD*HWSZ + hw;
    const float* xh1p = xh + (1*NB + b)*HD*HWSZ + hw;

    // prefetch next tile's bases (global_prefetch_b8)
    int tn = t + nw;
    if (tn < NTILES){
      int bn2 = (tn*16) / HWSZ, hwn = (tn*16) - bn2*HWSZ;
      __builtin_prefetch(xf + bn2*HD*HWSZ + hwn, 0, 1);
      __builtin_prefetch(xh + bn2*HD*HWSZ + hwn, 0, 1);
      __builtin_prefetch(xp + bn2*HD*HWSZ + hwn, 0, 1);
    }

    float ha1 = h_att[(1*NB + b)*HWSZ + hw + q];
    float ha2 = h_att[(2*NB + b)*HWSZ + hw + q];
    float pau = p_att[(1*NB + b)*HWSZ + hw + q] + p_att[(2*NB + b)*HWSZ + hw + q]
              + p_att[(3*NB + b)*HWSZ + hw + q] + p_att[(4*NB + b)*HWSZ + hw + q];
    float pal = p_att[(5*NB + b)*HWSZ + hw + q] + p_att[(6*NB + b)*HWSZ + hw + q];

    Frag B;
    // Decomposition (shared d-weights): cat(xf*h_att, xh half)
    B = buildB_gg(xfp, ha1, xh0p, 1.f, q, hi);
    v8f mu = applyBlock(fragL, biasL, scr, B, 0,1,2, 0,1, lane);
    B = buildB_gg(xfp, ha2, xh1p, 1.f, q, hi);
    v8f ml = applyBlock(fragL, biasL, scr, B, 0,1,2, 0,1, lane);
    // Composition upper: 4 parts
#pragma unroll
    for (int i = 0; i < 4; ++i){
      const float* xpp = xp + (i*NB + b)*HD*HWSZ + hw;
      B = buildB_gg(xh0p, 1.f, xpp, pau, q, hi);
      v8f d = applyBlock(fragL, biasL, scr, B, 3,4,5, 2,3, lane);
      mu += d;
    }
    // Composition lower: 2 parts
#pragma unroll
    for (int i = 4; i < 6; ++i){
      const float* xpp = xp + (i*NB + b)*HD*HWSZ + hw;
      B = buildB_gg(xh1p, 1.f, xpp, pal, q, hi);
      v8f d = applyBlock(fragL, biasL, scr, B, 6,7,8, 4,5, lane);
      ml += d;
    }
    // ConvGRU cells -> final hidden states
    gruCell(fragL, biasL, scr, mu, xh0p, 9,10,11, 6,7,
            out + (0*NB + b)*HD*HWSZ + hw, lane);
    gruCell(fragL, biasL, scr, ml, xh1p, 12,13,14, 8,9,
            out + (1*NB + b)*HD*HWSZ + hw, lane);
  }
}

// ---------------------------------------------------------------------------
extern "C" void kernel_launch(void* const* d_in, const int* in_sizes, int n_in,
                              void* d_out, int out_size, void* d_ws, size_t ws_size,
                              hipStream_t stream)
{
  (void)in_sizes; (void)n_in; (void)out_size; (void)ws_size;
  const float* xf   = (const float*)d_in[0];
  const float* xh   = (const float*)d_in[1];
  const float* xp   = (const float*)d_in[2];
  const float* hatt = (const float*)d_in[3];
  const float* patt = (const float*)d_in[4];

  prep_weights<<<(WS_DW + 255)/256, 256, 0, stream>>>(
      (const float*)d_in[5],  (const float*)d_in[6],  (const float*)d_in[7],  (const float*)d_in[8],
      (const float*)d_in[9],  (const float*)d_in[10], (const float*)d_in[11], (const float*)d_in[12],
      (const float*)d_in[13], (const float*)d_in[14], (const float*)d_in[15], (const float*)d_in[16],
      (const float*)d_in[17], (const float*)d_in[18], (const float*)d_in[19], (const float*)d_in[20],
      (const float*)d_in[21], (const float*)d_in[22], (const float*)d_in[23], (const float*)d_in[24],
      (unsigned*)d_ws);

  half_graph_main<<<768, 256, 0, stream>>>(xf, xh, xp, hatt, patt,
      (const unsigned*)d_ws, (float*)d_out);
}
